// Block_8985071583801
// MI455X (gfx1250) — compile-verified, matrix operand order checked by reading, not run on hardware
//
#include <hip/hip_runtime.h>
#include <hip/hip_bf16.h>
#include <math.h>

typedef __attribute__((ext_vector_type(16))) _Float16 v16h;
typedef __attribute__((ext_vector_type(8)))  _Float16 v8h;
typedef __attribute__((ext_vector_type(8)))  float    v8f;
typedef int v4i __attribute__((__vector_size__(16)));   // matches builtin param type

#define BM 128
#define BN 128
#define GK 64     // K depth staged per iteration (two WMMA k-steps)
#define LK 72     // padded LDS row stride in halves (144B, 16B-aligned)

#define AS1 __attribute__((address_space(1)))
#define AS3 __attribute__((address_space(3)))

// ---- CDNA5 async global->LDS path (guarded; falls back to sync staging) ----
#if defined(__has_builtin)
#  if __has_builtin(__builtin_amdgcn_global_load_async_to_lds_b128)
#    define HAVE_ASYNC_LDS 1
#  endif
#endif
#ifndef HAVE_ASYNC_LDS
#  define HAVE_ASYNC_LDS 0
#endif

#if HAVE_ASYNC_LDS
#  if __has_builtin(__builtin_amdgcn_s_wait_asynccnt)
#    define WAIT_ASYNC0() __builtin_amdgcn_s_wait_asynccnt(0)
#  else
#    define WAIT_ASYNC0() asm volatile("s_wait_asynccnt 0x0" ::: "memory")
#  endif
#endif

// -------------------- generic f32 -> f16 copy --------------------
__global__ void k_f32_to_f16(const float* __restrict__ src, _Float16* __restrict__ dst, int n) {
    int i = blockIdx.x * blockDim.x + threadIdx.x;
    if (i < n) dst[i] = (_Float16)src[i];
}

// ffn_conv_w [O=1024][I=1024][3][3] f32  ->  9 matrices Wk[p][O][I] f16 (p = kh*3+kw)
__global__ void k_conv_w_split(const float* __restrict__ src, _Float16* __restrict__ dst) {
    int i = blockIdx.x * blockDim.x + threadIdx.x;        // over 9*1024*1024
    if (i >= 9 * 1024 * 1024) return;
    int p = i >> 20;
    int o = (i >> 10) & 1023;
    int c = i & 1023;
    int kh = p / 3, kw = p % 3;
    dst[i] = (_Float16)src[(((size_t)o * 1024 + c) * 3 + kh) * 3 + kw];
}

// -------------------- WMMA tiled GEMM --------------------
__device__ inline v16h ld_frag(const _Float16* p) {
    v8h lo = *(const v8h*)(p);
    v8h hi = *(const v8h*)(p + 16);
    return __builtin_shufflevector(lo, hi, 0,1,2,3,4,5,6,7,8,9,10,11,12,13,14,15);
}

// C[8192][N] = A[8192][K] (f32, row stride lda, optional (dh,dw) spatial shift) * Wh[N][K] f16
// EPI: 0 none, 1 softplus(f+bias), 2 exact GELU, 3 relu6(f*scale+bias)
template<bool SHIFT, int EPI, bool ACC, bool CHW>
__global__ void __launch_bounds__(256)
k_gemm(const float* __restrict__ A, int lda,
       const _Float16* __restrict__ Wh,
       float* __restrict__ C, int ldc,
       int N, int K, int dh, int dw,
       const float* __restrict__ bias,
       const float* __restrict__ scale)
{
    __shared__ __align__(16) _Float16 As[BM * LK];
    __shared__ __align__(16) _Float16 Bs[BN * LK];

    const int tid    = threadIdx.x;
    const int lane   = tid & 31;
    const int wid    = tid >> 5;
    const int wave_m = wid >> 2;   // 0..1  -> 64 rows
    const int wave_n = wid & 3;    // 0..3  -> 32 cols
    const int mBase  = blockIdx.x * BM;
    const int nBase  = blockIdx.y * BN;

    // ---- K-invariant staging maps (hoisted out of the K loop) ----
    int  aRow[8];  bool aVal[8];           // 8 passes of 16 rows, 16 threads/row
    const int aCol = (tid & 15) * 4;
    #pragma unroll
    for (int rr = 0; rr < 8; ++rr) {
        int row = rr * 16 + (tid >> 4);
        int m = mBase + row;
        if (SHIFT) {
            int b = m >> 10, hw = m & 1023;
            int h = (hw >> 5) + dh, w = (hw & 31) + dw;
            aVal[rr] = (h >= 0) && (h < 32) && (w >= 0) && (w < 32);
            aRow[rr] = (b << 10) + (h << 5) + w;
        } else { aVal[rr] = true; aRow[rr] = m; }
    }
    int  bRow[4];  bool bVal[4];           // 4 passes of 32 rows, 8 threads/row
    const int bCol = (tid & 7) * 8;
    #pragma unroll
    for (int rr = 0; rr < 4; ++rr) {
        bRow[rr] = nBase + rr * 32 + (tid >> 3);
        bVal[rr] = bRow[rr] < N;
    }

    v8f acc[4][2];
    #pragma unroll
    for (int i = 0; i < 4; ++i)
        #pragma unroll
        for (int j = 0; j < 2; ++j)
            acc[i][j] = {};

    const int lrow  = lane & 15;
    const int khalf = lane >> 4;

    for (int k0 = 0; k0 < K; k0 += GK) {
#if HAVE_ASYNC_LDS
        // ---- B tile: fire-and-forget async global->LDS (ASYNCcnt) ----
        #pragma unroll
        for (int rr = 0; rr < 4; ++rr) {
            if (bVal[rr]) {
                __builtin_amdgcn_global_load_async_to_lds_b128(
                    (AS1 v4i*)(Wh + (size_t)bRow[rr] * K + k0 + bCol),
                    (AS3 v4i*)(&Bs[(rr * 32 + (tid >> 3)) * LK + bCol]),
                    0, 0);
            }
        }
#endif
        // ---- A tile: batch-issue global loads into registers (needs f32->f16) ----
        float4 av[8];
        #pragma unroll
        for (int rr = 0; rr < 8; ++rr) {
            av[rr] = make_float4(0.f, 0.f, 0.f, 0.f);
            if (aVal[rr])
                av[rr] = *(const float4*)(A + (size_t)aRow[rr] * lda + k0 + aCol);
        }
#if !HAVE_ASYNC_LDS
        v8h bv[4];
        #pragma unroll
        for (int rr = 0; rr < 4; ++rr) {
            bv[rr] = {};
            if (bVal[rr])
                bv[rr] = *(const v8h*)(Wh + (size_t)bRow[rr] * K + k0 + bCol);
        }
#endif
        // prefetch next weight tile into L2 (global_prefetch_b8)
        if (k0 + GK < K) {
            int n = nBase + (tid >> 1);
            if (n < N) __builtin_prefetch(Wh + (size_t)n * K + k0 + GK, 0, 1);
        }
        // ---- convert + store A to LDS (overlaps with in-flight async B copies) ----
        #pragma unroll
        for (int rr = 0; rr < 8; ++rr) {
            _Float16* d = &As[(rr * 16 + (tid >> 4)) * LK + aCol];
            d[0] = (_Float16)av[rr].x; d[1] = (_Float16)av[rr].y;
            d[2] = (_Float16)av[rr].z; d[3] = (_Float16)av[rr].w;
        }
#if HAVE_ASYNC_LDS
        WAIT_ASYNC0();
#else
        #pragma unroll
        for (int rr = 0; rr < 4; ++rr)
            *(v8h*)(&Bs[(rr * 32 + (tid >> 3)) * LK + bCol]) = bv[rr];
#endif
        __syncthreads();

        // ---- two 16x16x32 k-steps per staged tile: 16 WMMAs ----
        #pragma unroll
        for (int ks = 0; ks < 2; ++ks) {
            const int c0 = ks * 32;
            v16h af[4], bf[2];
            #pragma unroll
            for (int mt = 0; mt < 4; ++mt)
                af[mt] = ld_frag(&As[(wave_m * 64 + mt * 16 + lrow) * LK + c0 + khalf * 8]);
            #pragma unroll
            for (int nt = 0; nt < 2; ++nt)
                bf[nt] = ld_frag(&Bs[(wave_n * 32 + nt * 16 + lrow) * LK + c0 + khalf * 8]);
            #pragma unroll
            for (int mt = 0; mt < 4; ++mt)
                #pragma unroll
                for (int nt = 0; nt < 2; ++nt)
                    acc[mt][nt] = __builtin_amdgcn_wmma_f32_16x16x32_f16(
                        false, af[mt], false, bf[nt], (short)0, acc[mt][nt], false, false);
        }
        __syncthreads();
    }

    // ---- epilogue ----
    const int mhalf = lane >> 4;
    #pragma unroll
    for (int mt = 0; mt < 4; ++mt) {
        #pragma unroll
        for (int nt = 0; nt < 2; ++nt) {
            #pragma unroll
            for (int r = 0; r < 8; ++r) {
                int m = mBase + wave_m * 64 + mt * 16 + mhalf * 8 + r;
                int n = nBase + wave_n * 32 + nt * 16 + lrow;
                if (n >= N) continue;
                float f = acc[mt][nt][r];
                size_t idx;
                if (CHW) {
                    int b = m >> 10, hw = m & 1023;
                    idx = ((size_t)b * N + n) * 1024 + hw;
                } else {
                    idx = (size_t)m * ldc + n;
                }
                if (ACC) f += C[idx];
                if (EPI == 1) {
                    f += bias[n];
                    f = (f > 20.f) ? f : log1pf(__expf(f));
                } else if (EPI == 2) {
                    f = 0.5f * f * (1.f + erff(f * 0.70710678f));
                } else if (EPI == 3) {
                    f = f * scale[n] + bias[n];
                    f = fminf(fmaxf(f, 0.f), 6.f);
                }
                C[idx] = f;
            }
        }
    }
}

// -------------------- front-end kernels --------------------
__global__ void k_x_to_seq(const float* __restrict__ x, float* __restrict__ seq) {
    int i = blockIdx.x * blockDim.x + threadIdx.x;         // 8*512*1024
    if (i >= 8 * 512 * 1024) return;
    int b = i >> 19, c = (i >> 10) & 511, hw = i & 1023;
    seq[((size_t)(b * 1024 + hw)) * 1024 + c] = x[i];
}

__global__ void k_conv1x1_b0(const float* __restrict__ x, const float* __restrict__ wp,
                             const float* __restrict__ sc, const float* __restrict__ bi,
                             float* __restrict__ t0) {
    int i = blockIdx.x * blockDim.x + threadIdx.x;         // 8*128*1024
    if (i >= 8 * 128 * 1024) return;
    int b = i >> 17, o = (i >> 10) & 127, hw = i & 1023;
    float s = 0.f;
    const float* xr = x + ((size_t)b * 512) * 1024 + hw;
    const float* wr = wp + (size_t)o * 512;
    for (int c = 0; c < 512; ++c) s += xr[(size_t)c * 1024] * wr[c];
    s = s * sc[o] + bi[o];
    t0[i] = fminf(fmaxf(s, 0.f), 6.f);
}

__global__ void k_mean_bcast(const float* __restrict__ t0, float* __restrict__ seq) {
    __shared__ float red[256];
    int bo = blockIdx.x;
    int b = bo >> 7, o = bo & 127;
    const float* row = t0 + (size_t)bo * 1024;
    float s = 0.f;
    for (int l = threadIdx.x; l < 1024; l += 256) s += row[l];
    red[threadIdx.x] = s;
    __syncthreads();
    for (int st = 128; st > 0; st >>= 1) {
        if (threadIdx.x < st) red[threadIdx.x] += red[threadIdx.x + st];
        __syncthreads();
    }
    float mean = red[0] * (1.f / 1024.f);
    for (int l = threadIdx.x; l < 1024; l += 256)
        seq[((size_t)(b * 1024 + l)) * 1024 + 512 + o] = mean;
}

__global__ void k_adaptive_pool(const float* __restrict__ x, float* __restrict__ pooled, int p) {
    int i = blockIdx.x * blockDim.x + threadIdx.x;         // 8*512*p*p
    int total = 8 * 512 * p * p;
    if (i >= total) return;
    int pp = p * p;
    int bc = i / pp, q = i % pp;
    int iy = q / p, ix = q % p;
    int sh = (iy * 32) / p, eh = ((iy + 1) * 32 + p - 1) / p;
    int sw = (ix * 32) / p, ew = ((ix + 1) * 32 + p - 1) / p;
    float s = 0.f;
    const float* xr = x + (size_t)bc * 1024;
    for (int h = sh; h < eh; ++h)
        for (int w = sw; w < ew; ++w)
            s += xr[h * 32 + w];
    pooled[i] = s / (float)((eh - sh) * (ew - sw));
}

__global__ void k_conv1x1_pool(const float* __restrict__ pooled, const float* __restrict__ wp,
                               const float* __restrict__ sc, const float* __restrict__ bi,
                               float* __restrict__ tp, int p) {
    int pp = p * p;
    int i = blockIdx.x * blockDim.x + threadIdx.x;
    if (i >= 8 * 128 * pp) return;
    int b = i / (128 * pp), rem = i % (128 * pp);
    int o = rem / pp, q = rem % pp;
    float s = 0.f;
    const float* pr = pooled + ((size_t)b * 512) * pp + q;
    const float* wr = wp + (size_t)o * 512;
    for (int c = 0; c < 512; ++c) s += pr[(size_t)c * pp] * wr[c];
    s = s * sc[o] + bi[o];
    tp[i] = fminf(fmaxf(s, 0.f), 6.f);
}

__global__ void k_bilinear_seq(const float* __restrict__ tp, float* __restrict__ seq,
                               int p, int branch) {
    int i = blockIdx.x * blockDim.x + threadIdx.x;         // 8*128*1024
    if (i >= 8 * 128 * 1024) return;
    int b = i >> 17, o = (i >> 10) & 127, hw = i & 1023;
    int h = hw >> 5, w = hw & 31;
    float fy = ((float)h + 0.5f) * (float)p / 32.f - 0.5f;
    float fx = ((float)w + 0.5f) * (float)p / 32.f - 0.5f;
    fy = fminf(fmaxf(fy, 0.f), (float)(p - 1));
    fx = fminf(fmaxf(fx, 0.f), (float)(p - 1));
    int y0 = (int)fy, x0 = (int)fx;
    int y1 = min(y0 + 1, p - 1), x1 = min(x0 + 1, p - 1);
    float wy = fy - (float)y0, wx = fx - (float)x0;
    const float* t = tp + ((size_t)(b * 128 + o)) * p * p;
    float v = (1.f - wy) * ((1.f - wx) * t[y0 * p + x0] + wx * t[y0 * p + x1]) +
              wy        * ((1.f - wx) * t[y1 * p + x0] + wx * t[y1 * p + x1]);
    seq[((size_t)(b * 1024 + hw)) * 1024 + 512 + 128 * branch + o] = v;
}

// -------------------- mamba elementwise --------------------
__global__ void k_conv1d_silu(const float* __restrict__ xz, const float* __restrict__ cw,
                              const float* __restrict__ cb, float* __restrict__ xm) {
    int i = blockIdx.x * blockDim.x + threadIdx.x;         // 8192*2048
    if (i >= 8192 * 2048) return;
    int m = i >> 11, d = i & 2047;
    int b = m >> 10, l = m & 1023;
    float s = cb[d];
    #pragma unroll
    for (int k = 0; k < 4; ++k) {
        int ls = l - 3 + k;
        if (ls >= 0)
            s += xz[((size_t)((b << 10) + ls)) * 4096 + d] * cw[d * 4 + k];
    }
    xm[i] = s / (1.f + __expf(-s));
}

__global__ void __launch_bounds__(256)
k_scan(const float* __restrict__ u, const float* __restrict__ dt,
       const float* __restrict__ xdbl, const float* __restrict__ A_log,
       const float* __restrict__ Dp, const float* __restrict__ xz,
       float* __restrict__ ys)
{
    __shared__ float sB[16], sC[16];
    int b = blockIdx.y;
    int d = blockIdx.x * 256 + threadIdx.x;
    float a[16], h[16];
    #pragma unroll
    for (int n = 0; n < 16; ++n) { a[n] = -__expf(A_log[d * 16 + n]); h[n] = 0.f; }
    float Dd = Dp[d];
    for (int l = 0; l < 1024; ++l) {
        int m = (b << 10) + l;
        if (threadIdx.x < 32) {
            int t = threadIdx.x;
            float v = xdbl[(size_t)m * 96 + 64 + t];
            if (t < 16) sB[t] = v; else sC[t - 16] = v;
        }
        __syncthreads();
        float ut  = u [(size_t)m * 2048 + d];
        float dtt = dt[(size_t)m * 2048 + d];
        float du  = dtt * ut;
        float y = 0.f;
        #pragma unroll
        for (int n = 0; n < 16; ++n) {
            h[n] = __expf(dtt * a[n]) * h[n] + du * sB[n];
            y += h[n] * sC[n];
        }
        float z = xz[(size_t)m * 4096 + 2048 + d];
        float sz = z / (1.f + __expf(-z));
        ys[(size_t)m * 2048 + d] = (y + ut * Dd) * sz;
        __syncthreads();
    }
}

// -------------------- host side --------------------
static inline int cdiv(int a, int b) { return (a + b - 1) / b; }

extern "C" void kernel_launch(void* const* d_in, const int* in_sizes, int n_in,
                              void* d_out, int out_size, void* d_ws, size_t ws_size,
                              hipStream_t stream) {
    const float* x        = (const float*)d_in[0];
    const float* w_pool   = (const float*)d_in[1];
    const float* pbn_s    = (const float*)d_in[2];
    const float* pbn_b    = (const float*)d_in[3];
    const float* in_proj  = (const float*)d_in[4];
    const float* c1w      = (const float*)d_in[5];
    const float* c1b      = (const float*)d_in[6];
    const float* x_projw  = (const float*)d_in[7];
    const float* dt_projw = (const float*)d_in[8];
    const float* dt_projb = (const float*)d_in[9];
    const float* A_log    = (const float*)d_in[10];
    const float* D_param  = (const float*)d_in[11];
    const float* out_projw= (const float*)d_in[12];
    const float* ffn_w    = (const float*)d_in[13];
    const float* ffn_s    = (const float*)d_in[14];
    const float* ffn_b    = (const float*)d_in[15];
    const float* fc1w     = (const float*)d_in[16];
    const float* fc2w     = (const float*)d_in[17];
    float* out = (float*)d_out;

    char* ws = (char*)d_ws;
    size_t cur = 0;
    auto alloc = [&](size_t bytes) -> char* {
        char* p = ws + cur;
        cur += (bytes + 255) & ~(size_t)255;
        return p;
    };
    float* seq     = (float*)alloc(8192ull * 1024 * 4);
    float* xz      = (float*)alloc(8192ull * 4096 * 4);
    float* xm      = (float*)alloc(8192ull * 2048 * 4);
    float* xdbl    = (float*)alloc(8192ull * 96 * 4);
    float* dtb     = (float*)alloc(8192ull * 2048 * 4);
    float* ys      = (float*)alloc(8192ull * 2048 * 4);
    float* outtok  = (float*)alloc(8192ull * 1024 * 4);
    float* convout = (float*)alloc(8192ull * 1024 * 4);
    float* fc1out  = (float*)alloc(8192ull * 512 * 4);
    float* t0      = (float*)alloc(8ull * 128 * 1024 * 4);
    float* pool5   = (float*)alloc(8ull * 512 * 25 * 4);
    float* pool9   = (float*)alloc(8ull * 512 * 81 * 4);
    float* pool13  = (float*)alloc(8ull * 512 * 169 * 4);
    float* tp5     = (float*)alloc(8ull * 128 * 25 * 4);
    float* tp9     = (float*)alloc(8ull * 128 * 81 * 4);
    float* tp13    = (float*)alloc(8ull * 128 * 169 * 4);
    _Float16* hInP = (_Float16*)alloc(4096ull * 1024 * 2);
    _Float16* hXP  = (_Float16*)alloc(96ull * 2048 * 2);
    _Float16* hDT  = (_Float16*)alloc(2048ull * 64 * 2);
    _Float16* hOutP= (_Float16*)alloc(1024ull * 2048 * 2);
    _Float16* hCw9 = (_Float16*)alloc(9ull * 1024 * 1024 * 2);
    _Float16* hFc1 = (_Float16*)alloc(512ull * 1024 * 2);
    _Float16* hFc2 = (_Float16*)alloc(128ull * 512 * 2);

    const int T = 256;
    k_f32_to_f16<<<cdiv(4096 * 1024, T), T, 0, stream>>>(in_proj, hInP, 4096 * 1024);
    k_f32_to_f16<<<cdiv(96 * 2048, T),  T, 0, stream>>>(x_projw, hXP, 96 * 2048);
    k_f32_to_f16<<<cdiv(2048 * 64, T),  T, 0, stream>>>(dt_projw, hDT, 2048 * 64);
    k_f32_to_f16<<<cdiv(1024 * 2048, T),T, 0, stream>>>(out_projw, hOutP, 1024 * 2048);
    k_f32_to_f16<<<cdiv(512 * 1024, T), T, 0, stream>>>(fc1w, hFc1, 512 * 1024);
    k_f32_to_f16<<<cdiv(128 * 512, T),  T, 0, stream>>>(fc2w, hFc2, 128 * 512);
    k_conv_w_split<<<cdiv(9 * 1024 * 1024, T), T, 0, stream>>>(ffn_w, hCw9);

    k_x_to_seq<<<cdiv(8 * 512 * 1024, T), T, 0, stream>>>(x, seq);
    k_conv1x1_b0<<<cdiv(8 * 128 * 1024, T), T, 0, stream>>>(x, w_pool, pbn_s, pbn_b, t0);
    k_mean_bcast<<<8 * 128, 256, 0, stream>>>(t0, seq);
    const int ps[3] = {5, 9, 13};
    float* pools[3] = {pool5, pool9, pool13};
    float* tps[3]   = {tp5, tp9, tp13};
    for (int i = 0; i < 3; ++i) {
        int p = ps[i], pp = p * p;
        k_adaptive_pool<<<cdiv(8 * 512 * pp, T), T, 0, stream>>>(x, pools[i], p);
        k_conv1x1_pool<<<cdiv(8 * 128 * pp, T), T, 0, stream>>>(
            pools[i], w_pool + (size_t)(i + 1) * 128 * 512,
            pbn_s + (i + 1) * 128, pbn_b + (i + 1) * 128, tps[i], p);
        k_bilinear_seq<<<cdiv(8 * 128 * 1024, T), T, 0, stream>>>(tps[i], seq, p, i + 1);
    }

    dim3 blk(256);
    // in_proj
    { dim3 g(64, cdiv(4096, BN));
      k_gemm<false,0,false,false><<<g, blk, 0, stream>>>(seq, 1024, hInP, xz, 4096,
                                                         4096, 1024, 0, 0, nullptr, nullptr); }
    k_conv1d_silu<<<cdiv(8192 * 2048, T), T, 0, stream>>>(xz, c1w, c1b, xm);
    // x_proj
    { dim3 g(64, cdiv(96, BN));
      k_gemm<false,0,false,false><<<g, blk, 0, stream>>>(xm, 2048, hXP, xdbl, 96,
                                                         96, 2048, 0, 0, nullptr, nullptr); }
    // dt_proj + softplus
    { dim3 g(64, cdiv(2048, BN));
      k_gemm<false,1,false,false><<<g, blk, 0, stream>>>(xdbl, 96, hDT, dtb, 2048,
                                                         2048, 64, 0, 0, dt_projb, nullptr); }
    // selective scan
    { dim3 g(8, 8);
      k_scan<<<g, 256, 0, stream>>>(xm, dtb, xdbl, A_log, D_param, xz, ys); }
    // out_proj
    { dim3 g(64, cdiv(1024, BN));
      k_gemm<false,0,false,false><<<g, blk, 0, stream>>>(ys, 2048, hOutP, outtok, 1024,
                                                         1024, 2048, 0, 0, nullptr, nullptr); }
    // ConvFFN 3x3 as 9 shifted implicit-GEMM passes
    for (int pass = 0; pass < 9; ++pass) {
        int kh = pass / 3, kw = pass % 3;
        dim3 g(64, cdiv(1024, BN));
        const _Float16* Wk = hCw9 + (size_t)pass * 1024 * 1024;
        if (pass == 0)
            k_gemm<true,0,false,false><<<g, blk, 0, stream>>>(outtok, 1024, Wk, convout, 1024,
                                                              1024, 1024, kh - 1, kw - 1, nullptr, nullptr);
        else if (pass < 8)
            k_gemm<true,0,true,false><<<g, blk, 0, stream>>>(outtok, 1024, Wk, convout, 1024,
                                                             1024, 1024, kh - 1, kw - 1, nullptr, nullptr);
        else
            k_gemm<true,3,true,false><<<g, blk, 0, stream>>>(outtok, 1024, Wk, convout, 1024,
                                                             1024, 1024, kh - 1, kw - 1, ffn_b, ffn_s);
    }
    // fc1 + GELU
    { dim3 g(64, cdiv(512, BN));
      k_gemm<false,2,false,false><<<g, blk, 0, stream>>>(convout, 1024, hFc1, fc1out, 512,
                                                         512, 1024, 0, 0, nullptr, nullptr); }
    // fc2 -> d_out in [B,128,H,W]
    { dim3 g(64, cdiv(128, BN));
      k_gemm<false,0,false,true><<<g, blk, 0, stream>>>(fc1out, 512, hFc2, out, 128,
                                                        128, 512, 0, 0, nullptr, nullptr); }
    (void)in_sizes; (void)n_in; (void)out_size; (void)ws_size;
}
